// Projector_41609643164336
// MI455X (gfx1250) — matrix operand order repaired and checked by reading, not compiled, not run on hardware
//
#include <hip/hip_runtime.h>

// ---------------- problem constants ----------------
#define BB   32
#define CC   384
#define LL   3136
#define HI_  56
#define LK   784
#define HK   28
#define NH   12
#define DH   32
#define LQP  3200   // L padded to 25*128 for GEMM tiling (pad cols zeroed)
#define LKP  896    // Lk padded to 7*128
#define LKS  800    // Lk padded to 25*32 for kv K-loop (pad zeroed in softmax)

typedef __bf16 bf16;
typedef __attribute__((ext_vector_type(16))) __bf16 v16bf;
typedef __attribute__((ext_vector_type(8)))  float  v8f;

struct __align__(16) U4 { unsigned int x, y, z, w; };
struct __align__(16) F4 { float v[4]; };
union Frag  { U4 q[2]; v16bf v; };              // 32B = 16 bf16 = one WMMA operand frag/lane
union Pack8 { U4 u; bf16 h[8]; };
union P8U   { U4 u4; unsigned short s[8]; };    // raw bf16 bit access for dword packing

__device__ __forceinline__ float gelu_f(float v) {
    return 0.5f * v * (1.0f + erff(v * 0.70710678118654752440f));
}
__device__ __forceinline__ float wsum(float v) {
#pragma unroll
    for (int o = 16; o > 0; o >>= 1) v += __shfl_xor(v, o, 32);
    return v;
}
__device__ __forceinline__ float wmax(float v) {
#pragma unroll
    for (int o = 16; o > 0; o >>= 1) v = fmaxf(v, __shfl_xor(v, o, 32));
    return v;
}

// ---------------- depthwise 3x3 stride 1 (q branch) + BN stats ----------------
__global__ __launch_bounds__(256) void conv_dw_s1(
    const float* __restrict__ x, const float* __restrict__ dwk,
    bf16* __restrict__ yq, float* __restrict__ cstats)
{
    const int c = blockIdx.x, b = blockIdx.y;
    const float* xp = x + ((size_t)b * CC + c) * LL;
    bf16* op = yq + ((size_t)b * CC + c) * LQP;
    float kw[9];
#pragma unroll
    for (int i = 0; i < 9; ++i) kw[i] = dwk[i * CC + c];   // branch 0
    float s1 = 0.f, s2 = 0.f;
    for (int p = threadIdx.x; p < LQP; p += 256) {
        float acc = 0.f;
        if (p < LL) {
            const int oy = p / HI_, ox = p - oy * HI_;
#pragma unroll
            for (int dy = 0; dy < 3; ++dy) {
                const int iy = oy + dy - 1;
                if (iy < 0 || iy >= HI_) continue;
#pragma unroll
                for (int dx = 0; dx < 3; ++dx) {
                    const int ix = ox + dx - 1;
                    if (ix < 0 || ix >= HI_) continue;
                    acc += xp[iy * HI_ + ix] * kw[dy * 3 + dx];
                }
            }
            s1 += acc; s2 += acc * acc;
        }
        op[p] = (bf16)acc;               // pad cols get zeros
    }
    __shared__ float r1[8], r2[8];
    s1 = wsum(s1); s2 = wsum(s2);
    const int lane = threadIdx.x & 31, wv = threadIdx.x >> 5;
    if (lane == 0) { r1[wv] = s1; r2[wv] = s2; }
    __syncthreads();
    if (threadIdx.x == 0) {
        float a = 0.f, q2 = 0.f;
        for (int i = 0; i < 8; ++i) { a += r1[i]; q2 += r2[i]; }
        atomicAdd(&cstats[c], a);
        atomicAdd(&cstats[CC + c], q2);
    }
}

// ---------------- depthwise 3x3 stride 2 (k,v branches) + BN stats ----------------
__global__ __launch_bounds__(256) void conv_dw_s2(
    const float* __restrict__ x, const float* __restrict__ dwk,
    bf16* __restrict__ yk, bf16* __restrict__ yv, float* __restrict__ cstats)
{
    const int c = blockIdx.x, b = blockIdx.y, br = 1 + blockIdx.z;
    const float* xp = x + ((size_t)b * CC + c) * LL;
    bf16* op = (br == 1 ? yk : yv) + ((size_t)b * CC + c) * LKP;
    float kw[9];
#pragma unroll
    for (int i = 0; i < 9; ++i) kw[i] = dwk[(br * 9 + i) * CC + c];
    float s1 = 0.f, s2 = 0.f;
    for (int p = threadIdx.x; p < LKP; p += 256) {
        float acc = 0.f;
        if (p < LK) {
            const int oy = p / HK, ox = p - oy * HK;
#pragma unroll
            for (int dy = 0; dy < 3; ++dy) {
                const int iy = 2 * oy + dy - 1;
                if (iy < 0 || iy >= HI_) continue;
#pragma unroll
                for (int dx = 0; dx < 3; ++dx) {
                    const int ix = 2 * ox + dx - 1;
                    if (ix < 0 || ix >= HI_) continue;
                    acc += xp[iy * HI_ + ix] * kw[dy * 3 + dx];
                }
            }
            s1 += acc; s2 += acc * acc;
        }
        op[p] = (bf16)acc;
    }
    __shared__ float r1[8], r2[8];
    s1 = wsum(s1); s2 = wsum(s2);
    const int lane = threadIdx.x & 31, wv = threadIdx.x >> 5;
    if (lane == 0) { r1[wv] = s1; r2[wv] = s2; }
    __syncthreads();
    if (threadIdx.x == 0) {
        float a = 0.f, q2 = 0.f;
        for (int i = 0; i < 8; ++i) { a += r1[i]; q2 += r2[i]; }
        atomicAdd(&cstats[br * 2 * CC + c], a);
        atomicAdd(&cstats[br * 2 * CC + CC + c], q2);
    }
}

// ---------------- fold BN2 into pointwise weights/bias, cast weights to bf16 ----------------
__global__ __launch_bounds__(128) void fold_bn(
    const float* __restrict__ cstats, const float* __restrict__ g2,
    const float* __restrict__ b2, const float* __restrict__ pww,
    const float* __restrict__ pwb, bf16* __restrict__ wf, float* __restrict__ bfold)
{
    const int o = blockIdx.x, br = blockIdx.y;
    const float icnt = (br == 0) ? 1.0f / (float)(BB * LL) : 1.0f / (float)(BB * LK);
    const float* st = cstats + (size_t)br * 2 * CC;
    float part = 0.f;
    for (int c = threadIdx.x; c < CC; c += 128) {
        const float mu  = st[c] * icnt;
        const float var = st[CC + c] * icnt - mu * mu;
        const float s   = g2[br * CC + c] * rsqrtf(var + 1e-5f);
        const float t   = b2[br * CC + c] - mu * s;
        const float w   = pww[((size_t)br * CC + o) * CC + c];
        wf[((size_t)br * CC + o) * CC + c] = (bf16)(w * s);
        part += w * t;
    }
    part = wsum(part);
    __shared__ float red[4];
    const int lane = threadIdx.x & 31, wv = threadIdx.x >> 5;
    if (lane == 0) red[wv] = part;
    __syncthreads();
    if (threadIdx.x == 0)
        bfold[br * CC + o] = pwb[br * CC + o] + red[0] + red[1] + red[2] + red[3];
}

// ---------------- pointwise 1x1 conv as WMMA GEMM: D = GELU(W'·Y + b') --------------
// Block tile 64(M=out chan) x 128(N=token); 8 waves = 2x4; wave tile 32x32 (4 accums).
// Double-buffered LDS, register-staged next tile, prefetch of tile after next.
// MODE 0: q -> fused per-head softmax over DH, bf16 out [B][L][C] (token-major)
// MODE 1: k -> f32 out [B][C][LKP]    MODE 2: v -> bf16 out [B][C][LKP]
template <int MODE>
__global__ __launch_bounds__(256) void gemm_pw(
    const bf16* __restrict__ Wp, const float* __restrict__ bias,
    const bf16* __restrict__ Y, const int Lpad,
    float* __restrict__ outF, bf16* __restrict__ outH)
{
    const int b = blockIdx.z;
    const int o_blk = blockIdx.x * 64;
    const int l_blk = blockIdx.y * 128;
    const int tid = threadIdx.x, lane = tid & 31, wv = tid >> 5;
    const int wm = wv >> 2, wn = wv & 3;
    const int lm = lane & 15, hi = lane >> 4;

    __shared__ __align__(16) bf16 As[2][64 * 40];   // pitch 40 elems (80B): 16B-aligned chunks
    __shared__ __align__(16) bf16 Bs[2][128 * 40];  // transposed tile [l][k], k-pairs dword-packed

    const bf16* Yb = Y + (size_t)b * CC * Lpad;

    // staging coordinates
    const int ar  = tid >> 2;            // A row 0..63
    const int ach = (tid & 3) << 3;      // A k-chunk {0,8,16,24}
    const int bkp = tid >> 4;            // B k dword col 0..15 (k = 2*bkp, 2*bkp+1)
    const int blc = (tid & 15) << 3;     // B l base 0..120 (8 l's)

    const bf16* aSrc  = Wp + (size_t)(o_blk + ar) * CC + ach;
    const bf16* bSrc0 = Yb + (size_t)(2 * bkp) * Lpad + l_blk + blc;
    const bf16* bSrc1 = bSrc0 + Lpad;

    // preload + commit K-tile 0
    U4 aReg; P8U bReg0, bReg1;
    aReg     = *(const U4*)(aSrc);
    bReg0.u4 = *(const U4*)(bSrc0);
    bReg1.u4 = *(const U4*)(bSrc1);
    *(U4*)&As[0][ar * 40 + ach] = aReg;
    {
        unsigned int* bs32 = (unsigned int*)&Bs[0][0];
#pragma unroll
        for (int i = 0; i < 8; ++i)
            bs32[(blc + i) * 20 + bkp] =
                (unsigned int)bReg0.s[i] | ((unsigned int)bReg1.s[i] << 16);
    }
    __syncthreads();

    v8f acc[2][2];
    { v8f z = {}; acc[0][0] = z; acc[0][1] = z; acc[1][0] = z; acc[1][1] = z; }

    for (int kk = 0; kk < CC; kk += 32) {
        const int  cur  = (kk >> 5) & 1;
        const bool more = (kk + 32) < CC;
        if (more) {          // register-stage next K tile; prefetch the one after
            aReg     = *(const U4*)(aSrc + kk + 32);
            bReg0.u4 = *(const U4*)(bSrc0 + (size_t)(kk + 32) * Lpad);
            bReg1.u4 = *(const U4*)(bSrc1 + (size_t)(kk + 32) * Lpad);
            if (kk + 64 < CC) {
                __builtin_prefetch(aSrc + kk + 64, 0, 1);
                __builtin_prefetch(bSrc0 + (size_t)(kk + 64) * Lpad, 0, 1);
            }
        }
        Frag af[2], bfr[2];
#pragma unroll
        for (int mi = 0; mi < 2; ++mi) {
            const bf16* ap = &As[cur][(wm * 32 + mi * 16 + lm) * 40];
            af[mi].q[0] = *(const U4*)(ap + hi * 16);
            af[mi].q[1] = *(const U4*)(ap + hi * 16 + 8);
        }
#pragma unroll
        for (int ni = 0; ni < 2; ++ni) {
            const bf16* bp = &Bs[cur][(wn * 32 + ni * 16 + lm) * 40 + hi * 16];
            bfr[ni].q[0] = *(const U4*)(bp);
            bfr[ni].q[1] = *(const U4*)(bp + 8);
        }
#pragma unroll
        for (int mi = 0; mi < 2; ++mi)
#pragma unroll
            for (int ni = 0; ni < 2; ++ni)
                acc[mi][ni] = __builtin_amdgcn_wmma_f32_16x16x32_bf16(
                    false, af[mi].v, false, bfr[ni].v, (short)0, acc[mi][ni], false, false);
        if (more) {          // commit staged tile to the alternate buffer
            *(U4*)&As[cur ^ 1][ar * 40 + ach] = aReg;
            unsigned int* bs32 = (unsigned int*)&Bs[cur ^ 1][0];
#pragma unroll
            for (int i = 0; i < 8; ++i)
                bs32[(blc + i) * 20 + bkp] =
                    (unsigned int)bReg0.s[i] | ((unsigned int)bReg1.s[i] << 16);
        }
        __syncthreads();
    }

    if (MODE == 0) {
        // wave covers o range [head_base, head_base+32) == exactly one head.
        // lane + lane^16 together hold all 32 per-head features of token l.
        const int head_base = o_blk + wm * 32;
#pragma unroll
        for (int ni = 0; ni < 2; ++ni) {
            const int l = l_blk + wn * 32 + ni * 16 + lm;
            float gv[2][8];
            float mx = -3.4e38f;
#pragma unroll
            for (int mi = 0; mi < 2; ++mi)
#pragma unroll
                for (int j = 0; j < 8; ++j) {
                    const int o = head_base + mi * 16 + hi * 8 + j;
                    const float v = gelu_f(acc[mi][ni][j] + bias[o]);
                    gv[mi][j] = v;
                    mx = fmaxf(mx, v);
                }
            mx = fmaxf(mx, __shfl_xor(mx, 16, 32));
            float sm = 0.f;
#pragma unroll
            for (int mi = 0; mi < 2; ++mi)
#pragma unroll
                for (int j = 0; j < 8; ++j) {
                    const float e = expf(gv[mi][j] - mx);
                    gv[mi][j] = e;
                    sm += e;
                }
            sm += __shfl_xor(sm, 16, 32);
            const float inv = 1.0f / sm;
            if (l < LL) {
#pragma unroll
                for (int mi = 0; mi < 2; ++mi) {
                    Pack8 pk;
#pragma unroll
                    for (int j = 0; j < 8; ++j) pk.h[j] = (bf16)(gv[mi][j] * inv);
                    *(U4*)&outH[((size_t)b * LL + l) * CC + head_base + mi * 16 + hi * 8] = pk.u;
                }
            }
        }
    } else {
#pragma unroll
        for (int mi = 0; mi < 2; ++mi)
#pragma unroll
            for (int ni = 0; ni < 2; ++ni) {
                const int l = l_blk + wn * 32 + ni * 16 + lm;
#pragma unroll
                for (int j = 0; j < 8; ++j) {
                    const int o = o_blk + wm * 32 + mi * 16 + hi * 8 + j;
                    const float v = gelu_f(acc[mi][ni][j] + bias[o]);
                    if (MODE == 1) outF[((size_t)b * CC + o) * Lpad + l] = v;
                    else           outH[((size_t)b * CC + o) * Lpad + l] = (bf16)v;
                }
            }
    }
}

// ---------------- k softmax over tokens (per b,c row of 784) ----------------
__global__ __launch_bounds__(256) void k_softmax(
    const float* __restrict__ kin, bf16* __restrict__ ks)
{
    const int c = blockIdx.x, b = blockIdx.y;
    const float* row = kin + ((size_t)b * CC + c) * LKP;
    bf16* orow = ks + ((size_t)b * CC + c) * LKS;
    __shared__ float buf[LK];
    __shared__ float redm[8], reds[8];
    const int lane = threadIdx.x & 31, wv = threadIdx.x >> 5;
    float mx = -3.4e38f;
    for (int p = threadIdx.x; p < LK; p += 256) {
        const float v = row[p];
        buf[p] = v;
        mx = fmaxf(mx, v);
    }
    mx = wmax(mx);
    if (lane == 0) redm[wv] = mx;
    __syncthreads();
    float m = -3.4e38f;
    for (int i = 0; i < 8; ++i) m = fmaxf(m, redm[i]);
    float sm = 0.f;
    for (int p = threadIdx.x; p < LK; p += 256) {
        const float e = expf(buf[p] - m);
        buf[p] = e;
        sm += e;
    }
    sm = wsum(sm);
    if (lane == 0) reds[wv] = sm;
    __syncthreads();
    float tot = 0.f;
    for (int i = 0; i < 8; ++i) tot += reds[i];
    const float inv = 1.0f / tot;
    for (int p = threadIdx.x; p < LKS; p += 256)
        orow[p] = (p < LK) ? (bf16)(buf[p] * inv) : (bf16)0.0f;   // zero pad -> safe WMMA tail
}

// ---------------- kv = ks^T(dxtok) @ v(tokxe), per (b,h); direct-global WMMA frags --------
__global__ __launch_bounds__(64) void kv_gemm(
    const bf16* __restrict__ ks, const bf16* __restrict__ vb, bf16* __restrict__ kvT)
{
    const int h = blockIdx.x, b = blockIdx.y;
    const int lane = threadIdx.x & 31, wv = threadIdx.x >> 5;
    const int lm = lane & 15, hi = lane >> 4;
    const bf16* ap = ks + ((size_t)b * CC + h * DH + wv * 16 + lm) * LKS;
    v8f acc[2];
    { v8f z = {}; acc[0] = z; acc[1] = z; }
    for (int t0 = 0; t0 < LKS; t0 += 32) {
        Frag a;
        a.q[0] = *(const U4*)(ap + t0 + hi * 16);
        a.q[1] = *(const U4*)(ap + t0 + hi * 16 + 8);
#pragma unroll
        for (int et = 0; et < 2; ++et) {
            const bf16* bp = vb + ((size_t)b * CC + h * DH + et * 16 + lm) * LKP + t0 + hi * 16;
            Frag bfr;
            bfr.q[0] = *(const U4*)(bp);
            bfr.q[1] = *(const U4*)(bp + 8);
            acc[et] = __builtin_amdgcn_wmma_f32_16x16x32_bf16(
                false, a.v, false, bfr.v, (short)0, acc[et], false, false);
        }
    }
#pragma unroll
    for (int et = 0; et < 2; ++et) {   // store transposed [e][d] so att B-frags are K-contiguous
        const int e = et * 16 + lm;
        Pack8 pk;
#pragma unroll
        for (int j = 0; j < 8; ++j) pk.h[j] = (bf16)acc[et][j];
        *(U4*)&kvT[(((size_t)b * NH + h) * DH + e) * DH + wv * 16 + hi * 8] = pk.u;
    }
}

// ---------------- att = qs @ kv (K=32), fused weighted residual + BN1 stats ----------------
__global__ __launch_bounds__(256) void att_residual(
    const bf16* __restrict__ qs, const bf16* __restrict__ kvT,
    const float* __restrict__ x, const float* __restrict__ addw,
    float* __restrict__ ypre, float* __restrict__ bstats)
{
    const int h = blockIdx.x, b = blockIdx.y;
    const int tid = threadIdx.x, lane = tid & 31, wv = tid >> 5;
    const int lm = lane & 15, hi = lane >> 4;
    const float w0 = fmaxf(addw[0], 0.f), w1 = fmaxf(addw[1], 0.f);
    const float winv = 1.0f / (w0 + w1 + 1e-12f);
    const float wt0 = w0 * winv, wt1 = w1 * winv;

    Frag bfr[2];                      // kv is loop-invariant: hoist B-frags
#pragma unroll
    for (int et = 0; et < 2; ++et) {
        const bf16* bp = kvT + (((size_t)b * NH + h) * DH + et * 16 + lm) * DH + hi * 16;
        bfr[et].q[0] = *(const U4*)(bp);
        bfr[et].q[1] = *(const U4*)(bp + 8);
    }
    float ls1[2] = {0.f, 0.f}, ls2[2] = {0.f, 0.f};
    for (int tile = wv; tile < LL / 16; tile += 8) {   // 196 tiles, exact
        const int l0 = tile * 16;
        if (tile + 8 < LL / 16) {     // prefetch next tile for this wave
            __builtin_prefetch(qs + ((size_t)b * LL + l0 + 128 + lm) * CC + h * DH, 0, 1);
            __builtin_prefetch(x + ((size_t)b * CC + h * DH + lm) * LL + l0 + 128, 0, 1);
        }
        const bf16* ap = qs + ((size_t)b * LL + l0 + lm) * CC + h * DH;
        Frag a;
        a.q[0] = *(const U4*)(ap + hi * 16);
        a.q[1] = *(const U4*)(ap + hi * 16 + 8);
#pragma unroll
        for (int et = 0; et < 2; ++et) {
            v8f z = {};
            const v8f d = __builtin_amdgcn_wmma_f32_16x16x32_bf16(
                false, a.v, false, bfr[et].v, (short)0, z, false, false);
            const int c = h * DH + et * 16 + lm;       // channel fixed per lane
            const size_t base = ((size_t)b * CC + c) * LL + l0 + hi * 8;
            const F4 x0 = *(const F4*)(x + base);
            const F4 x1 = *(const F4*)(x + base + 4);
            F4 o0, o1;
#pragma unroll
            for (int j = 0; j < 4; ++j) {
                const float y0 = wt0 * d[j]     + wt1 * x0.v[j];
                const float y1 = wt0 * d[4 + j] + wt1 * x1.v[j];
                o0.v[j] = y0; o1.v[j] = y1;
                ls1[et] += y0 + y1;
                ls2[et] += y0 * y0 + y1 * y1;
            }
            *(F4*)(ypre + base)     = o0;
            *(F4*)(ypre + base + 4) = o1;
        }
    }
    __shared__ float sb[64];
    if (tid < 64) sb[tid] = 0.f;
    __syncthreads();
#pragma unroll
    for (int et = 0; et < 2; ++et) {
        atomicAdd(&sb[(et * 16 + lm) * 2 + 0], ls1[et]);
        atomicAdd(&sb[(et * 16 + lm) * 2 + 1], ls2[et]);
    }
    __syncthreads();
    if (tid < DH) {
        atomicAdd(&bstats[h * DH + tid],      sb[tid * 2 + 0]);
        atomicAdd(&bstats[CC + h * DH + tid], sb[tid * 2 + 1]);
    }
}

// ---------------- final BatchNorm1d normalize ----------------
__global__ __launch_bounds__(256) void final_bn(
    const float* __restrict__ ypre, const float* __restrict__ bstats,
    const float* __restrict__ g1, const float* __restrict__ b1,
    float* __restrict__ out)
{
    const size_t base = ((size_t)blockIdx.x * 256 + threadIdx.x) * 4;
    const int c = (int)((base / LL) % CC);
    const float invn = 1.0f / (float)(BB * LL);
    const float mu  = bstats[c] * invn;
    const float var = bstats[CC + c] * invn - mu * mu;
    const float sc  = g1[c] * rsqrtf(var + 1e-5f);
    const float sh  = b1[c] - mu * sc;
    const F4 v = *(const F4*)(ypre + base);
    F4 o;
#pragma unroll
    for (int j = 0; j < 4; ++j) o.v[j] = v.v[j] * sc + sh;
    *(F4*)(out + base) = o;
}

// ---------------- launcher ----------------
extern "C" void kernel_launch(void* const* d_in, const int* in_sizes, int n_in,
                              void* d_out, int out_size, void* d_ws, size_t ws_size,
                              hipStream_t stream) {
    (void)in_sizes; (void)n_in; (void)out_size; (void)ws_size;
    const float* x   = (const float*)d_in[0];
    const float* dwk = (const float*)d_in[1];
    const float* g2  = (const float*)d_in[2];
    const float* b2  = (const float*)d_in[3];
    const float* pww = (const float*)d_in[4];
    const float* pwb = (const float*)d_in[5];
    const float* aw  = (const float*)d_in[6];
    const float* g1  = (const float*)d_in[7];
    const float* b1  = (const float*)d_in[8];
    float* out = (float*)d_out;

    char* p = (char*)d_ws;
    auto alloc = [&](size_t bytes) -> char* {
        char* r = p;
        p += (bytes + 255) & ~(size_t)255;
        return r;
    };
    bf16*  yq    = (bf16*) alloc((size_t)BB * CC * LQP * 2);
    bf16*  yk    = (bf16*) alloc((size_t)BB * CC * LKP * 2);
    bf16*  yv    = (bf16*) alloc((size_t)BB * CC * LKP * 2);
    bf16*  qsb   = (bf16*) alloc((size_t)BB * LL  * CC  * 2);
    float* kq    = (float*)alloc((size_t)BB * CC * LKP * 4);
    bf16*  vb    = (bf16*) alloc((size_t)BB * CC * LKP * 2);
    bf16*  ksb   = (bf16*) alloc((size_t)BB * CC * LKS * 2);
    bf16*  kvT   = (bf16*) alloc((size_t)BB * NH * DH * DH * 2);
    float* ypre  = (float*)alloc((size_t)BB * CC * LL * 4);
    bf16*  wf    = (bf16*) alloc((size_t)3 * CC * CC * 2);
    float* bfold = (float*)alloc((size_t)3 * CC * 4);
    float* cst   = (float*)alloc((size_t)3 * 2 * CC * 4);
    float* bst   = (float*)alloc((size_t)2 * CC * 4);

    hipMemsetAsync(cst, 0, (size_t)3 * 2 * CC * 4, stream);
    hipMemsetAsync(bst, 0, (size_t)2 * CC * 4, stream);

    conv_dw_s1<<<dim3(CC, BB),    256, 0, stream>>>(x, dwk, yq, cst);
    conv_dw_s2<<<dim3(CC, BB, 2), 256, 0, stream>>>(x, dwk, yk, yv, cst);
    fold_bn   <<<dim3(CC, 3),     128, 0, stream>>>(cst, g2, b2, pww, pwb, wf, bfold);

    gemm_pw<0><<<dim3(CC / 64, LQP / 128, BB), 256, 0, stream>>>(wf,               bfold,          yq, LQP, nullptr, qsb);
    gemm_pw<1><<<dim3(CC / 64, LKP / 128, BB), 256, 0, stream>>>(wf + CC * CC,     bfold + CC,     yk, LKP, kq,      nullptr);
    gemm_pw<2><<<dim3(CC / 64, LKP / 128, BB), 256, 0, stream>>>(wf + 2 * CC * CC, bfold + 2 * CC, yv, LKP, nullptr, vb);

    k_softmax   <<<dim3(CC, BB), 256, 0, stream>>>(kq, ksb);
    kv_gemm     <<<dim3(NH, BB),  64, 0, stream>>>(ksb, vb, kvT);
    att_residual<<<dim3(NH, BB), 256, 0, stream>>>(qsb, kvT, x, aw, ypre, bst);
    final_bn    <<<(BB * CC * LL / 4) / 256, 256, 0, stream>>>(ypre, bst, g1, b1, out);
}
